// toyVFdisc_84593675862401
// MI455X (gfx1250) — matrix operand order repaired and checked by reading, not compiled, not run on hardware
//
#include <hip/hip_runtime.h>
#include <hip/hip_bf16.h>
#include <math.h>

typedef __attribute__((ext_vector_type(8)))  __bf16 v8bf;
typedef __attribute__((ext_vector_type(16))) __bf16 v16bf;
typedef __attribute__((ext_vector_type(8)))  float  v8f;

constexpr int BATCH  = 1024;
constexpr int NS0    = 512;
constexpr int NS1    = 2048;
constexpr int NCAT   = NS0 + NS1;   // 2560 (output/state width)
constexpr int KDATA  = 1024;        // S2
constexpr int TSTEPS = 30;

// ---------------------------------------------------------------------------
// Pack Wcat (K = NCAT, N = NCAT) into B-matrix tile layout, hi/lo bf16 split.
// B element (k, n) = Wcat[n][k]; Wcat rows 0..511 = [w0 | w1], 512.. = [w5 | w3].
// Packed layout: tile (kt = k/32, nt = n/16); lane = n%16 + 16*((k%32/8)&1);
// within-lane element j = k%8 + 8*(k%32/16); 16 contiguous bf16 per lane.
// ---------------------------------------------------------------------------
__global__ __launch_bounds__(256) void pack_wcat_kernel(
    const float* __restrict__ w0, const float* __restrict__ w1,
    const float* __restrict__ w3, const float* __restrict__ w5,
    __bf16* __restrict__ Bh, __bf16* __restrict__ Bl) {
  int idx = blockIdx.x * 256 + threadIdx.x;        // over NCAT*NCAT
  int k = idx / NCAT, n = idx % NCAT;
  float v;
  if (n < NS0) v = (k < NS0) ? w0[n * NS0 + k] : w1[n * NS1 + (k - NS0)];
  else         v = (k < NS0) ? w5[(n - NS0) * NS0 + k]
                             : w3[(size_t)(n - NS0) * NS1 + (k - NS0)];
  int kt = k >> 5, kr = k & 31, nt = n >> 4, nr = n & 15;
  int lane = ((kr >> 3) & 1) ? nr + 16 : nr;
  int j    = (kr & 7) + ((kr >> 4) << 3);
  size_t off = (((size_t)kt * (NCAT / 16) + nt) * 32 + lane) * 16 + j;
  __bf16 h = (__bf16)v;
  Bh[off] = h;
  Bl[off] = (__bf16)(v - (float)h);
}

// Same packing for W2cat (K = KDATA, N = NCAT): rows 0..511 = w2, 512.. = w4.
__global__ __launch_bounds__(256) void pack_w2cat_kernel(
    const float* __restrict__ w2, const float* __restrict__ w4,
    __bf16* __restrict__ Bh, __bf16* __restrict__ Bl) {
  int idx = blockIdx.x * 256 + threadIdx.x;        // over KDATA*NCAT
  int k = idx / NCAT, n = idx % NCAT;
  float v = (n < NS0) ? w2[n * KDATA + k] : w4[(n - NS0) * KDATA + k];
  int kt = k >> 5, kr = k & 31, nt = n >> 4, nr = n & 15;
  int lane = ((kr >> 3) & 1) ? nr + 16 : nr;
  int j    = (kr & 7) + ((kr >> 4) << 3);
  size_t off = (((size_t)kt * (NCAT / 16) + nt) * 32 + lane) * 16 + j;
  __bf16 h = (__bf16)v;
  Bh[off] = h;
  Bl[off] = (__bf16)(v - (float)h);
}

// X0 = [s0 | s1] row-major [BATCH, NCAT], hi/lo bf16 split.
__global__ __launch_bounds__(256) void pack_x0_kernel(
    const float* __restrict__ s0, const float* __restrict__ s1,
    __bf16* __restrict__ Xh, __bf16* __restrict__ Xl) {
  int idx = blockIdx.x * 256 + threadIdx.x;        // over BATCH*NCAT
  int m = idx / NCAT, n = idx % NCAT;
  float v = (n < NS0) ? s0[m * NS0 + n] : s1[(size_t)m * NS1 + (n - NS0)];
  __bf16 h = (__bf16)v;
  Xh[idx] = h;
  Xl[idx] = (__bf16)(v - (float)h);
}

// Plain row-major fp32 -> bf16 hi/lo (used for `data`).
__global__ __launch_bounds__(256) void pack_rm_kernel(
    const float* __restrict__ src, __bf16* __restrict__ H, __bf16* __restrict__ L) {
  int idx = blockIdx.x * 256 + threadIdx.x;
  float v = src[idx];
  __bf16 h = (__bf16)v;
  H[idx] = h;
  L[idx] = (__bf16)(v - (float)h);
}

// ---------------------------------------------------------------------------
// A-fragment load for v_wmma_f32_16x16x32_bf16, 16-bit A layout (16x32):
// lanes 0-15 (row = base+lane)  : K = k0+0..7  (v0-3), k0+16..23 (v4-7)
// lanes 16-31 (row = base+lane-16): K = k0+8..15, k0+24..31
// => two contiguous 16-byte loads per lane from row-major A.
// ---------------------------------------------------------------------------
__device__ __forceinline__ v16bf load_a_frag(const __bf16* __restrict__ A,
                                             int K, int row, int kc, int sel) {
  const __bf16* base = A + (size_t)row * K + kc + sel;
  v8bf x = *(const v8bf*)(base);
  v8bf y = *(const v8bf*)(base + 16);
  return __builtin_shufflevector(x, y, 0, 1, 2, 3, 4, 5, 6, 7,
                                 8, 9, 10, 11, 12, 13, 14, 15);
}

// ---------------------------------------------------------------------------
// GEMM: out[M=1024, N=2560] = op( A[M,K] * B[K,N] + bias )
// A given as hi/lo bf16 row-major (stride K); B given hi/lo in packed tile
// layout. 3-term split product: Ah*Bh + Al*Bh + Ah*Bl, fp32 accumulate.
// Block = 256 threads = 8 waves (2 x 4); wave tile 32x32; WG tile 64x128.
// Epilogue: optional bias add, optional tanh, optional bf16 hi/lo store,
// optional fp32 row-major store, optional fp32 split (s0f,s1f) store.
// ---------------------------------------------------------------------------
__global__ __launch_bounds__(256) void gemm_wmma_kernel(
    const __bf16* __restrict__ Ah, const __bf16* __restrict__ Al, int K,
    const __bf16* __restrict__ Bh, const __bf16* __restrict__ Bl,
    const float* __restrict__ bias, int doTanh,
    __bf16* __restrict__ outH, __bf16* __restrict__ outL,
    float* __restrict__ outF, float* __restrict__ outSplit) {
  const int lane = threadIdx.x & 31;
  const int wave = threadIdx.x >> 5;
  const int m0 = blockIdx.y * 64 + (wave >> 2) * 32;
  const int n0 = blockIdx.x * 128 + (wave & 3) * 32;
  const int lrow = lane & 15;
  const int sel = (lane >> 4) * 8;
  const int ntn = NCAT / 16;

  v8f acc[2][2] = {};

  for (int kc = 0; kc < K; kc += 32) {
    v16bf aH[2], aL[2], bH[2], bL[2];
#pragma unroll
    for (int mi = 0; mi < 2; ++mi) {
      int row = m0 + mi * 16 + lrow;
      aH[mi] = load_a_frag(Ah, K, row, kc, sel);
      aL[mi] = load_a_frag(Al, K, row, kc, sel);
    }
#pragma unroll
    for (int ni = 0; ni < 2; ++ni) {
      size_t tile = (size_t)(kc >> 5) * ntn + (n0 >> 4) + ni;
      size_t base = (tile * 32 + lane) * 16;
      bH[ni] = *(const v16bf*)(Bh + base);
      bL[ni] = *(const v16bf*)(Bl + base);
    }
#pragma unroll
    for (int mi = 0; mi < 2; ++mi)
#pragma unroll
      for (int ni = 0; ni < 2; ++ni) {
        acc[mi][ni] = __builtin_amdgcn_wmma_f32_16x16x32_bf16(
            false, aH[mi], false, bH[ni], (short)0, acc[mi][ni], false, false);
        acc[mi][ni] = __builtin_amdgcn_wmma_f32_16x16x32_bf16(
            false, aL[mi], false, bH[ni], (short)0, acc[mi][ni], false, false);
        acc[mi][ni] = __builtin_amdgcn_wmma_f32_16x16x32_bf16(
            false, aH[mi], false, bL[ni], (short)0, acc[mi][ni], false, false);
      }
  }

  // Epilogue. C/D layout: VGPR r, lanes 0-15: (M = r,   N = lane)
  //                                lanes 16-31: (M = r+8, N = lane-16)
  const int colLane = lane & 15;
  const int rAdd = (lane >> 4) * 8;
#pragma unroll
  for (int mi = 0; mi < 2; ++mi)
#pragma unroll
    for (int ni = 0; ni < 2; ++ni) {
      int col = n0 + ni * 16 + colLane;
#pragma unroll
      for (int r = 0; r < 8; ++r) {
        int row = m0 + mi * 16 + rAdd + r;
        float v = acc[mi][ni][r];
        if (bias) v += bias[(size_t)row * NCAT + col];
        if (doTanh) v = tanhf(v);
        if (outH) {
          __bf16 h = (__bf16)v;
          outH[(size_t)row * NCAT + col] = h;
          outL[(size_t)row * NCAT + col] = (__bf16)(v - (float)h);
        }
        if (outF) outF[(size_t)row * NCAT + col] = v;
        if (outSplit) {
          if (col < NS0)
            outSplit[(size_t)row * NS0 + col] = v;
          else
            outSplit[(size_t)BATCH * NS0 + (size_t)row * NS1 + (col - NS0)] = v;
        }
      }
    }
}

extern "C" void kernel_launch(void* const* d_in, const int* in_sizes, int n_in,
                              void* d_out, int out_size, void* d_ws, size_t ws_size,
                              hipStream_t stream) {
  const float* data = (const float*)d_in[0];
  const float* s0   = (const float*)d_in[1];
  const float* s1   = (const float*)d_in[2];
  const float* w0   = (const float*)d_in[3];
  const float* w1   = (const float*)d_in[4];
  const float* w2   = (const float*)d_in[5];
  const float* w3   = (const float*)d_in[6];
  const float* w4   = (const float*)d_in[7];
  const float* w5   = (const float*)d_in[8];
  (void)w1; (void)in_sizes; (void)n_in; (void)out_size; (void)ws_size;

  // Workspace carve-out (256B aligned).
  char* p = (char*)d_ws;
  auto carve = [&](size_t bytes) {
    char* r = p;
    p += (bytes + 255) & ~(size_t)255;
    return r;
  };
  __bf16* WcatH = (__bf16*)carve((size_t)NCAT * NCAT * 2);
  __bf16* WcatL = (__bf16*)carve((size_t)NCAT * NCAT * 2);
  __bf16* W2H   = (__bf16*)carve((size_t)KDATA * NCAT * 2);
  __bf16* W2L   = (__bf16*)carve((size_t)KDATA * NCAT * 2);
  __bf16* DH    = (__bf16*)carve((size_t)BATCH * KDATA * 2);
  __bf16* DL    = (__bf16*)carve((size_t)BATCH * KDATA * 2);
  float*  Ccat  = (float*) carve((size_t)BATCH * NCAT * 4);
  __bf16* XH[2], *XL[2];
  XH[0] = (__bf16*)carve((size_t)BATCH * NCAT * 2);
  XL[0] = (__bf16*)carve((size_t)BATCH * NCAT * 2);
  XH[1] = (__bf16*)carve((size_t)BATCH * NCAT * 2);
  XL[1] = (__bf16*)carve((size_t)BATCH * NCAT * 2);

  dim3 blk(256);
  // 1) Pack weights into B-tile layout (hi/lo), state & data into row-major hi/lo.
  pack_wcat_kernel<<<(NCAT * NCAT) / 256, blk, 0, stream>>>(w0, w1, w3, w5, WcatH, WcatL);
  pack_w2cat_kernel<<<(KDATA * NCAT) / 256, blk, 0, stream>>>(w2, w4, W2H, W2L);
  pack_x0_kernel<<<(BATCH * NCAT) / 256, blk, 0, stream>>>(s0, s1, XH[0], XL[0]);
  pack_rm_kernel<<<(BATCH * KDATA) / 256, blk, 0, stream>>>(data, DH, DL);

  dim3 ggrid(NCAT / 128, BATCH / 64);
  // 2) Loop-invariant bias: Ccat = data @ [w2|w4]^T (fp32 row-major out).
  gemm_wmma_kernel<<<ggrid, blk, 0, stream>>>(DH, DL, KDATA, W2H, W2L,
                                              nullptr, 0, nullptr, nullptr,
                                              Ccat, nullptr);
  // 3) 30 fused Euler steps: X <- tanh(X @ Wcat^T + Ccat).
  for (int t = 0; t < TSTEPS; ++t) {
    int s = t & 1, d = s ^ 1;
    float* outSplit = (t == TSTEPS - 1) ? (float*)d_out : nullptr;
    gemm_wmma_kernel<<<ggrid, blk, 0, stream>>>(XH[s], XL[s], NCAT, WcatH, WcatL,
                                                Ccat, 1, XH[d], XL[d],
                                                nullptr, outSplit);
  }
}
